// Net_21646635172359
// MI455X (gfx1250) — matrix operand order repaired and checked by reading, not compiled, not run on hardware
//
#include <hip/hip_runtime.h>
#include <hip/hip_bf16.h>
#include <math.h>

#define NN   200000
#define EE   1600000
#define NN1  100000
#define EE2  100000
#define NN2  50000
#define BB   1000
#define C1K  32
#define C2K  64
#define HHID 25

typedef __attribute__((ext_vector_type(16))) _Float16 v16h;
typedef __attribute__((ext_vector_type(8)))  float    v8f;

union AF { v16h v; _Float16 h[16]; };

// ---------------------------------------------------------------- utilities
__global__ void k_zero(float* __restrict__ p, int n) {
  int t = blockIdx.x * blockDim.x + threadIdx.x;
  int stride = gridDim.x * blockDim.x;
  for (int i = t; i < n; i += stride) p[i] = 0.0f;
}

// transpose+pad n1_w2 [25,32] -> Wt1 [32 (N rows), 32 (K cols)] f16, zero pad K=25..31
__global__ void k_prep_w1t(const float* __restrict__ w2, _Float16* __restrict__ Wt1) {
  int t = blockIdx.x * blockDim.x + threadIdx.x;
  if (t >= 32 * 32) return;
  int n = t >> 5, k = t & 31;
  Wt1[n * 32 + k] = (k < HHID) ? (_Float16)w2[k * 32 + n] : (_Float16)0.0f;
}

// transpose+pad n2_w2 [25,2048] -> Wt2 [2048,32] f16
__global__ void k_prep_w2t(const float* __restrict__ w2, _Float16* __restrict__ Wt2) {
  int t = blockIdx.x * blockDim.x + threadIdx.x;
  if (t >= 2048 * 32) return;
  int c = t >> 5, k = t & 31;
  Wt2[c * 32 + k] = (k < HHID) ? (_Float16)w2[k * 2048 + c] : (_Float16)0.0f;
}

// transpose n2_b2 [32,64] (viewed as [i,o]) -> Bt2 [64,32] f16
__global__ void k_prep_b2t(const float* __restrict__ b2, _Float16* __restrict__ Bt2) {
  int t = blockIdx.x * blockDim.x + threadIdx.x;
  if (t >= 64 * 32) return;
  int o = t >> 5, i = t & 31;
  Bt2[o * 32 + i] = (_Float16)b2[i * 64 + o];
}

// -------------------------------------------------- layer-1 fused edge kernel
// 16 edges per single-wave workgroup.  hid = relu(ea@W1+b1) in scalar (K=2),
// msg = x[src] * (hid @ W2 + b2) via 2x v_wmma_f32_16x16x32_f16, atomic scatter.
__global__ __launch_bounds__(32) void k_edge1(
    const float* __restrict__ x, const int* __restrict__ ei,
    const float* __restrict__ ea, const float* __restrict__ w1,
    const float* __restrict__ b1, const _Float16* __restrict__ Wt1,
    const float* __restrict__ b2, float* __restrict__ agg1) {
  __shared__ __align__(16) _Float16 sh_hid[16][32];
  __shared__ float sh_x[16];
  __shared__ int   sh_dst[16];

  const int lane = threadIdx.x;
  const int e0 = blockIdx.x * 16;

  if (lane < 16) {
    int e = e0 + lane;
    float a0 = ea[2 * e], a1 = ea[2 * e + 1];
    int src = ei[e], dst = ei[EE + e];
    sh_x[lane] = x[src];
    sh_dst[lane] = dst;
#pragma unroll
    for (int h = 0; h < 32; ++h) {
      float v = 0.0f;
      if (h < HHID) {
        v = fmaf(a0, w1[h], fmaf(a1, w1[HHID + h], b1[h]));
        v = fmaxf(v, 0.0f);
      }
      sh_hid[lane][h] = (_Float16)v;
    }
  }
  __syncthreads();

  const int half = lane >> 4, n = lane & 15, m = n;
  // A fragment: 16x32 f16, lane holds row m, K range half*16..+15
  AF a;
  *(float4*)&a.h[0] = *(const float4*)&sh_hid[m][half * 16];
  *(float4*)&a.h[8] = *(const float4*)&sh_hid[m][half * 16 + 8];
  // B fragments: lane holds column n (resp. n+16), K range half*16..+15
  AF bf0, bf1;
  *(float4*)&bf0.h[0] = *(const float4*)&Wt1[n * 32 + half * 16];
  *(float4*)&bf0.h[8] = *(const float4*)&Wt1[n * 32 + half * 16 + 8];
  *(float4*)&bf1.h[0] = *(const float4*)&Wt1[(n + 16) * 32 + half * 16];
  *(float4*)&bf1.h[8] = *(const float4*)&Wt1[(n + 16) * 32 + half * 16 + 8];
  // bias via C operand (column-broadcast)
  v8f c0, c1;
  float bv0 = b2[n], bv1 = b2[n + 16];
#pragma unroll
  for (int r = 0; r < 8; ++r) { c0[r] = bv0; c1[r] = bv1; }

  v8f d0 = __builtin_amdgcn_wmma_f32_16x16x32_f16(false, a.v, false, bf0.v,
                                                  (short)0, c0, false, false);
  v8f d1 = __builtin_amdgcn_wmma_f32_16x16x32_f16(false, a.v, false, bf1.v,
                                                  (short)0, c1, false, false);
#pragma unroll
  for (int r = 0; r < 8; ++r) {
    int mm = r + half * 8;
    float xv = sh_x[mm];
    int dst = sh_dst[mm];
    atomicAdd(&agg1[dst * 32 + n],      xv * d0[r]);
    atomicAdd(&agg1[dst * 32 + n + 16], xv * d1[r]);
  }
}

// finalize layer1: h = elu(agg + x*root1 + bias1)  (in place on agg1)
__global__ void k_finalize1(float* __restrict__ agg1, const float* __restrict__ x,
                            const float* __restrict__ root1,
                            const float* __restrict__ bias1) {
  int t = blockIdx.x * blockDim.x + threadIdx.x;
  if (t >= NN * 32) return;
  int node = t >> 5, c = t & 31;
  float v = agg1[t] + x[node] * root1[c] + bias1[c];
  agg1[t] = v > 0.0f ? v : expm1f(v);
}

// pool1: pair-max feats, pair-mean pos (cluster1 = arange//2)
__global__ void k_pool1(const float* __restrict__ h1, const float* __restrict__ pos,
                        float* __restrict__ x1, float* __restrict__ pos1) {
  int t = blockIdx.x * blockDim.x + threadIdx.x;
  if (t >= NN1 * 32) return;
  int p = t >> 5, c = t & 31;
  x1[t] = fmaxf(h1[(2 * p) * 32 + c], h1[(2 * p + 1) * 32 + c]);
  if (c < 2) pos1[p * 2 + c] = 0.5f * (pos[(2 * p) * 2 + c] + pos[(2 * p + 1) * 2 + c]);
}

// global max |dpos| over edge_index2 (non-negative -> int-bits atomicMax)
__global__ void k_cartmax(const float* __restrict__ pos1, const int* __restrict__ ei2,
                          int* __restrict__ maxbits) {
  int e = blockIdx.x * blockDim.x + threadIdx.x;
  if (e >= EE2) return;
  int s = ei2[e], d = ei2[EE2 + e];
  float dx = fabsf(pos1[2 * s] - pos1[2 * d]);
  float dy = fabsf(pos1[2 * s + 1] - pos1[2 * d + 1]);
  atomicMax(maxbits, __float_as_int(fmaxf(dx, dy)));
}

// -------------------------------------------------- layer-2 fused edge kernel
// 16 edges per single-wave workgroup. Per tile: 4 bias-WMMA + 128 weight-WMMA
// (hid[16,32] @ Wt2 chunk), contracted with x1[src][i] in VALU, atomic scatter.
__global__ __launch_bounds__(32) void k_edge2(
    const float* __restrict__ x1, const float* __restrict__ pos1,
    const int* __restrict__ ei2, const float* __restrict__ w1,
    const float* __restrict__ b1, const _Float16* __restrict__ Wt2,
    const _Float16* __restrict__ Bt2, const int* __restrict__ maxbits,
    float* __restrict__ agg2) {
  __shared__ __align__(16) _Float16 sh_hid[16][32];
  __shared__ __align__(16) _Float16 sh_xh[16][32];
  __shared__ __align__(16) float    sh_xf[16][32];
  __shared__ int sh_dst[16];

  const int lane = threadIdx.x;
  const int e0 = blockIdx.x * 16;
  const float inv2m = 0.5f / __int_as_float(*maxbits);

  if (lane < 16) {
    int e = e0 + lane;
    int src = ei2[e], dst = ei2[EE2 + e];
    sh_dst[lane] = dst;
    float a0 = (pos1[2 * src]     - pos1[2 * dst])     * inv2m + 0.5f;
    float a1 = (pos1[2 * src + 1] - pos1[2 * dst + 1]) * inv2m + 0.5f;
#pragma unroll
    for (int h = 0; h < 32; ++h) {
      float v = 0.0f;
      if (h < HHID) {
        v = fmaf(a0, w1[h], fmaf(a1, w1[HHID + h], b1[h]));
        v = fmaxf(v, 0.0f);
      }
      sh_hid[lane][h] = (_Float16)v;
    }
    const float4* xr = (const float4*)(x1 + src * 32);
#pragma unroll
    for (int q = 0; q < 8; ++q) {
      float4 v4 = xr[q];
      ((float4*)&sh_xf[lane][0])[q] = v4;
      sh_xh[lane][4 * q + 0] = (_Float16)v4.x;
      sh_xh[lane][4 * q + 1] = (_Float16)v4.y;
      sh_xh[lane][4 * q + 2] = (_Float16)v4.z;
      sh_xh[lane][4 * q + 3] = (_Float16)v4.w;
    }
  }
  __syncthreads();

  const int half = lane >> 4, n = lane & 15, m = n;
  AF ah, ax;
  *(float4*)&ah.h[0] = *(const float4*)&sh_hid[m][half * 16];
  *(float4*)&ah.h[8] = *(const float4*)&sh_hid[m][half * 16 + 8];
  *(float4*)&ax.h[0] = *(const float4*)&sh_xh[m][half * 16];
  *(float4*)&ax.h[8] = *(const float4*)&sh_xh[m][half * 16 + 8];

  v8f zero = {};
#pragma unroll
  for (int jo = 0; jo < 4; ++jo) {
    // bias GEMM: acc = xs[16,32] @ b2[:, jo*16+n]
    AF bb;
    int ob = jo * 16 + n;
    *(float4*)&bb.h[0] = *(const float4*)&Bt2[ob * 32 + half * 16];
    *(float4*)&bb.h[8] = *(const float4*)&Bt2[ob * 32 + half * 16 + 8];
    v8f acc = __builtin_amdgcn_wmma_f32_16x16x32_f16(false, ax.v, false, bb.v,
                                                     (short)0, zero, false, false);
#pragma unroll 8
    for (int i = 0; i < 32; ++i) {
      int c = i * 64 + jo * 16 + n;   // column of the [16,2048] weight tile
      AF bf;
      *(float4*)&bf.h[0] = *(const float4*)&Wt2[c * 32 + half * 16];
      *(float4*)&bf.h[8] = *(const float4*)&Wt2[c * 32 + half * 16 + 8];
      v8f d = __builtin_amdgcn_wmma_f32_16x16x32_f16(false, ah.v, false, bf.v,
                                                     (short)0, zero, false, false);
#pragma unroll
      for (int r = 0; r < 8; ++r)
        acc[r] = fmaf(sh_xf[r + half * 8][i], d[r], acc[r]);
    }
#pragma unroll
    for (int r = 0; r < 8; ++r) {
      int mm = r + half * 8;
      atomicAdd(&agg2[sh_dst[mm] * 64 + jo * 16 + n], acc[r]);
    }
  }
}

// finalize layer2: h2 = elu(agg2 + x1@root2 + bias2)  (in place on agg2)
__global__ void k_finalize2(float* __restrict__ agg2, const float* __restrict__ x1,
                            const float* __restrict__ root2,
                            const float* __restrict__ bias2) {
  int t = blockIdx.x * blockDim.x + threadIdx.x;
  if (t >= NN1 * 64) return;
  int node = t >> 6, o = t & 63;
  float v = agg2[t] + bias2[o];
  const float* xr = x1 + node * 32;
#pragma unroll
  for (int i = 0; i < 32; ++i) v = fmaf(xr[i], root2[i * 64 + o], v);
  agg2[t] = v > 0.0f ? v : expm1f(v);
}

// pool2: pair-max (cluster2 = arange//2)
__global__ void k_pool2(const float* __restrict__ h2, float* __restrict__ x2) {
  int t = blockIdx.x * blockDim.x + threadIdx.x;
  if (t >= NN2 * 64) return;
  int q = t >> 6, o = t & 63;
  x2[t] = fmaxf(h2[(2 * q) * 64 + o], h2[(2 * q + 1) * 64 + o]);
}

// head: per graph (50 consecutive x2 rows) mean -> fc1+elu -> fc2 -> log_softmax
__global__ __launch_bounds__(64) void k_head(
    const float* __restrict__ x2, const float* __restrict__ fc1w,
    const float* __restrict__ fc1b, const float* __restrict__ fc2w,
    const float* __restrict__ fc2b, float* __restrict__ out) {
  __shared__ float sg[64], sz[128], sl[10];
  int b = blockIdx.x, t = threadIdx.x;
  float s = 0.0f;
  for (int r = 0; r < 50; ++r) s += x2[(b * 50 + r) * 64 + t];
  sg[t] = s * (1.0f / 50.0f);
  __syncthreads();
  for (int j = t; j < 128; j += 64) {
    float v = fc1b[j];
#pragma unroll 8
    for (int k = 0; k < 64; ++k) v = fmaf(sg[k], fc1w[k * 128 + j], v);
    sz[j] = v > 0.0f ? v : expm1f(v);
  }
  __syncthreads();
  if (t < 10) {
    float v = fc2b[t];
    for (int k = 0; k < 128; ++k) v = fmaf(sz[k], fc2w[k * 10 + t], v);
    sl[t] = v;
  }
  __syncthreads();
  if (t == 0) {
    float mx = sl[0];
    for (int j = 1; j < 10; ++j) mx = fmaxf(mx, sl[j]);
    float se = 0.0f;
    for (int j = 0; j < 10; ++j) se += expf(sl[j] - mx);
    float lse = mx + logf(se);
    for (int j = 0; j < 10; ++j) out[b * 10 + j] = sl[j] - lse;
  }
}

// ----------------------------------------------------------------- launcher
extern "C" void kernel_launch(void* const* d_in, const int* in_sizes, int n_in,
                              void* d_out, int out_size, void* d_ws, size_t ws_size,
                              hipStream_t stream) {
  const float* x      = (const float*)d_in[0];
  const float* pos    = (const float*)d_in[1];
  const int*   ei     = (const int*)  d_in[2];
  const float* ea     = (const float*)d_in[3];
  const int*   ei2    = (const int*)  d_in[6];
  const float* n1_w1  = (const float*)d_in[8];
  const float* n1_b1  = (const float*)d_in[9];
  const float* n1_w2  = (const float*)d_in[10];
  const float* n1_b2  = (const float*)d_in[11];
  const float* root1  = (const float*)d_in[12];
  const float* bias1  = (const float*)d_in[13];
  const float* n2_w1  = (const float*)d_in[14];
  const float* n2_b1  = (const float*)d_in[15];
  const float* n2_w2  = (const float*)d_in[16];
  const float* n2_b2  = (const float*)d_in[17];
  const float* root2  = (const float*)d_in[18];
  const float* bias2  = (const float*)d_in[19];
  const float* fc1w   = (const float*)d_in[20];
  const float* fc1b   = (const float*)d_in[21];
  const float* fc2w   = (const float*)d_in[22];
  const float* fc2b   = (const float*)d_in[23];
  float* out = (float*)d_out;

  char* w = (char*)d_ws;
  float* agg1 = (float*)w; w += (size_t)NN  * 32 * 4;
  float* x1v  = (float*)w; w += (size_t)NN1 * 32 * 4;
  float* pos1 = (float*)w; w += (size_t)NN1 * 2  * 4;
  float* agg2 = (float*)w; w += (size_t)NN1 * 64 * 4;
  float* x2v  = (float*)w; w += (size_t)NN2 * 64 * 4;
  _Float16* Wt1 = (_Float16*)w; w += 32 * 32 * 2;
  _Float16* Wt2 = (_Float16*)w; w += 2048 * 32 * 2;
  _Float16* Bt2 = (_Float16*)w; w += 64 * 32 * 2;
  int* maxbits = (int*)w; w += 16;

  // zero accumulators + max scalar
  k_zero<<<4096, 256, 0, stream>>>(agg1, NN * 32);
  k_zero<<<4096, 256, 0, stream>>>(agg2, NN1 * 64);
  k_zero<<<1, 32, 0, stream>>>((float*)maxbits, 1);

  // f16 transposed/padded weights
  k_prep_w1t<<<4, 256, 0, stream>>>(n1_w2, Wt1);
  k_prep_w2t<<<256, 256, 0, stream>>>(n2_w2, Wt2);
  k_prep_b2t<<<8, 256, 0, stream>>>(n2_b2, Bt2);

  // layer 1
  k_edge1<<<EE / 16, 32, 0, stream>>>(x, ei, ea, n1_w1, n1_b1, Wt1, n1_b2, agg1);
  k_finalize1<<<(NN * 32 + 255) / 256, 256, 0, stream>>>(agg1, x, root1, bias1);
  k_pool1<<<(NN1 * 32 + 255) / 256, 256, 0, stream>>>(agg1, pos, x1v, pos1);

  // cartesian scale for pooled graph
  k_cartmax<<<(EE2 + 255) / 256, 256, 0, stream>>>(pos1, ei2, maxbits);

  // layer 2
  k_edge2<<<EE2 / 16, 32, 0, stream>>>(x1v, pos1, ei2, n2_w1, n2_b1, Wt2, Bt2,
                                       maxbits, agg2);
  k_finalize2<<<(NN1 * 64 + 255) / 256, 256, 0, stream>>>(agg2, x1v, root2, bias2);
  k_pool2<<<(NN2 * 64 + 255) / 256, 256, 0, stream>>>(agg2, x2v);

  // head
  k_head<<<BB, 64, 0, stream>>>(x2v, fc1w, fc1b, fc2w, fc2b, out);

  (void)in_sizes; (void)n_in; (void)out_size; (void)ws_size;
}
